// VIMCO_Regressor_52286931862273
// MI455X (gfx1250) — compile-verified, matrix operand order checked by reading, not run on hardware
//
#include <hip/hip_runtime.h>
#include <math.h>

// ---------------------------------------------------------------------------
// Types for CDNA5 WMMA
// ---------------------------------------------------------------------------
typedef __attribute__((ext_vector_type(16))) __bf16 bf16x16;
typedef __attribute__((ext_vector_type(8)))  float  f32x8;
typedef __attribute__((ext_vector_type(2)))  float  f32x2;

union FragBF { uint4 u[2]; bf16x16 v; };

__device__ __forceinline__ unsigned short f2bf(float f) {
  union { float f; unsigned u; } x; x.f = f;
  unsigned r = x.u + 0x7fffu + ((x.u >> 16) & 1u);   // round-to-nearest-even
  return (unsigned short)(r >> 16);
}

__device__ __forceinline__ int imin2(int a, int b) { return a < b ? a : b; }
__device__ __forceinline__ int imax2(int a, int b) { return a > b ? a : b; }

// CDNA5 async global->LDS row copy (64 B), ASYNCcnt-tracked; offset applies to
// both LDS and global addresses per ISA pseudocode (CDNA5 ISA).
__device__ __forceinline__ void async_copy_row64B(unsigned lds_addr,
                                                  unsigned long long gaddr) {
#if defined(__AMDGCN__)
  asm volatile(
      "global_load_async_to_lds_b128 %0, %1, off\n\t"
      "global_load_async_to_lds_b128 %0, %1, off offset:16\n\t"
      "global_load_async_to_lds_b128 %0, %1, off offset:32\n\t"
      "global_load_async_to_lds_b128 %0, %1, off offset:48"
      :: "v"(lds_addr), "v"(gaddr) : "memory");
#endif
}

__device__ __forceinline__ void wait_async0() {
#if defined(__AMDGCN__)
  asm volatile("s_wait_asynccnt 0" ::: "memory");
#endif
}

// ---------------------------------------------------------------------------
// Kernel 1: x = concat(words, context) converted to bf16
// ---------------------------------------------------------------------------
__global__ void build_x_kernel(const float* __restrict__ words,
                               const float* __restrict__ ctx,
                               unsigned short* __restrict__ xb)
{
  int idx = blockIdx.x * blockDim.x + threadIdx.x;      // 256*4096 total
  int row = idx >> 12;
  int col = idx & 4095;
  float v = (col < 2048) ? words[(size_t)row * 2048 + col]
                         : ctx[(size_t)row * 2048 + (col - 2048)];
  xb[idx] = f2bf(v);
}

// ---------------------------------------------------------------------------
// Kernel 2/3: bf16 WMMA GEMM. A: bf16 (MxK, row-major, as u16) staged to LDS
// via CDNA5 async global->LDS copies. B: fp32 (KxN, row-major) converted to
// bf16 while staged to LDS as B^T (N-major).
// Block = 128 threads = 4 waves, each wave computes 64x64, block does 128x128.
// ACT: 0=none, 1=ELU.  OUT_BF16: 1 -> u16/bf16 output, 0 -> f32 output.
// ---------------------------------------------------------------------------
template<int ACT, int OUT_BF16>
__global__ __launch_bounds__(128)
void gemm_bf16_kernel(const unsigned short* __restrict__ A,
                      const float* __restrict__ B,
                      const float* __restrict__ bias,
                      void* __restrict__ Cout,
                      int M, int N, int K)
{
  __shared__ unsigned short Ash[128 * 40];   // 128 rows x 32 K, pad to 40
  __shared__ unsigned short Bsh[128 * 40];   // 128 cols (N) x 32 K, pad to 40

  const int tid  = threadIdx.x;
  const int wave = tid >> 5;
  const int lane = tid & 31;
  const int wm = (wave >> 1) * 64;
  const int wn = (wave & 1) * 64;
  const int m0 = blockIdx.y * 128;
  const int n0 = blockIdx.x * 128;
  const int hf = lane >> 4;       // lane half: selects K sub-blocks
  const int ll = lane & 15;

  const unsigned ashAddr = (unsigned)(uintptr_t)(&Ash[tid * 40]);
  const int kkB = tid >> 2;             // B staging: K-row handled by thread
  const int nbB = (tid & 3) * 32;       // B staging: first of 32 n's

  f32x8 acc[4][4] = {};

  for (int k0 = 0; k0 < K; k0 += 32) {
    // stage A tile: async DMA of one 32-element bf16 row (64 B) per thread,
    // no VGPR round-trip (GLOBAL_LOAD_ASYNC_TO_LDS_B128, ASYNCcnt)
    async_copy_row64B(ashAddr,
                      (unsigned long long)(uintptr_t)(A + (size_t)(m0 + tid) * K + k0));
    // stage B^T tile: thread t handles K-row kk, 32 consecutive n's (convert)
    {
      const float* src = B + (size_t)(k0 + kkB) * N + (n0 + nbB);
      #pragma unroll 8
      for (int ii = 0; ii < 32; ++ii)
        Bsh[(nbB + ii) * 40 + kkB] = f2bf(src[ii]);
      // pull next K-tile of the fp32 weight stream toward GL2
      if (k0 + 32 < K)
        __builtin_prefetch(src + (size_t)32 * N, 0, 1);
    }
    wait_async0();        // this wave's async LDS writes have landed
    __syncthreads();      // all waves' staging visible

    // fragments per ISA 16-bit A layout: lanes0-15 K={0..7,16..23},
    // lanes16-31 K={8..15,24..31}  -> two b128 LDS loads per fragment
    FragBF af[4], bfr[4];
    #pragma unroll
    for (int i = 0; i < 4; ++i) {
      const unsigned short* p = &Ash[(wm + i * 16 + ll) * 40 + 8 * hf];
      af[i].u[0] = *reinterpret_cast<const uint4*>(p);
      af[i].u[1] = *reinterpret_cast<const uint4*>(p + 16);
    }
    #pragma unroll
    for (int j = 0; j < 4; ++j) {
      const unsigned short* p = &Bsh[(wn + j * 16 + ll) * 40 + 8 * hf];
      bfr[j].u[0] = *reinterpret_cast<const uint4*>(p);
      bfr[j].u[1] = *reinterpret_cast<const uint4*>(p + 16);
    }
    #pragma unroll
    for (int i = 0; i < 4; ++i)
      #pragma unroll
      for (int j = 0; j < 4; ++j)
        acc[i][j] = __builtin_amdgcn_wmma_f32_16x16x32_bf16(
            false, af[i].v, false, bfr[j].v, (short)0, acc[i][j], false, false);
    __syncthreads();
  }

  // epilogue: C layout -> VGPR r: row = r + 8*(lane>=16), col = lane&15
  #pragma unroll
  for (int j = 0; j < 4; ++j) {
    const int col = n0 + wn + j * 16 + ll;
    const float bv = bias[col];
    #pragma unroll
    for (int i = 0; i < 4; ++i) {
      const int rowb = m0 + wm + i * 16 + hf * 8;
      #pragma unroll
      for (int r = 0; r < 8; ++r) {
        float v = acc[i][j][r] + bv;
        if (ACT == 1) v = (v > 0.f) ? v : (expf(v) - 1.f);    // ELU
        if (OUT_BF16)
          ((unsigned short*)Cout)[(size_t)(rowb + r) * N + col] = f2bf(v);
        else
          ((float*)Cout)[(size_t)(rowb + r) * N + col] = v;
      }
    }
  }
}

// ---------------------------------------------------------------------------
// Kernel 4: K = E E^T via fp32 WMMA (V_WMMA_F32_16X16X4_F32). Both operands
// are rows of E (B^T == E). Direct global loads; E (4 MB) lives in L2.
// ---------------------------------------------------------------------------
__global__ __launch_bounds__(128)
void syrk_f32_kernel(const float* __restrict__ E, float* __restrict__ Kout,
                     int n, int Kd)
{
#if defined(__AMDGCN__) && __has_builtin(__builtin_amdgcn_wmma_f32_16x16x4_f32)
  const int tid  = threadIdx.x;
  const int wave = tid >> 5;
  const int lane = tid & 31;
  const int wm = (wave >> 1) * 64;
  const int wn = (wave & 1) * 64;
  const int m0 = blockIdx.y * 128;
  const int n0 = blockIdx.x * 128;
  const int hf = lane >> 4;
  const int ll = lane & 15;

  f32x8 acc[4][4] = {};
  for (int k = 0; k < Kd; k += 4) {
    // fp32 A 16x4 layout: lanes0-15 K={0,1}, lanes16-31 K={2,3}
    f32x2 af[4], bfr[4];
    #pragma unroll
    for (int i = 0; i < 4; ++i)
      af[i] = *reinterpret_cast<const f32x2*>(
          E + (size_t)(m0 + wm + i * 16 + ll) * Kd + k + 2 * hf);
    #pragma unroll
    for (int j = 0; j < 4; ++j)
      bfr[j] = *reinterpret_cast<const f32x2*>(
          E + (size_t)(n0 + wn + j * 16 + ll) * Kd + k + 2 * hf);
    #pragma unroll
    for (int i = 0; i < 4; ++i)
      #pragma unroll
      for (int j = 0; j < 4; ++j)
        acc[i][j] = __builtin_amdgcn_wmma_f32_16x16x4_f32(
            false, af[i], false, bfr[j], (short)0, acc[i][j], false, false);
  }
  #pragma unroll
  for (int j = 0; j < 4; ++j) {
    const int col = n0 + wn + j * 16 + ll;
    #pragma unroll
    for (int i = 0; i < 4; ++i) {
      const int rowb = m0 + wm + i * 16 + hf * 8;
      #pragma unroll
      for (int r = 0; r < 8; ++r)
        Kout[(size_t)(rowb + r) * n + col] = acc[i][j][r];
    }
  }
#else
  // scalar fallback (also what the host pass parses)
  const int tid = threadIdx.x;
  const int m0 = blockIdx.y * 128;
  const int n0 = blockIdx.x * 128;
  for (int idx = tid; idx < 128 * 128; idx += 128) {
    int rr = idx >> 7, cc = idx & 127;
    int row = m0 + rr, col = n0 + cc;
    float s = 0.f;
    for (int k2 = 0; k2 < Kd; ++k2)
      s += E[(size_t)row * Kd + k2] * E[(size_t)col * Kd + k2];
    Kout[(size_t)row * n + col] = s;
  }
#endif
}

// ---------------------------------------------------------------------------
// Kernel 5: symmetric eigensolver, one workgroup, parallel cyclic Jacobi.
// 128 disjoint rotation pairs per round (round-robin tournament pairing).
// A is destroyed (diagonal -> eigenvalues), V accumulates eigenvectors (cols).
// ---------------------------------------------------------------------------
__device__ __forceinline__ void rr_pair(int r, int i, int n, int& p, int& q) {
  int a, b;
  if (i == 0) { a = r % (n - 1); b = n - 1; }
  else {
    a = (r + i) % (n - 1);
    b = (n - 1 - i + r) % (n - 1);
  }
  p = imin2(a, b); q = imax2(a, b);
}

__global__ __launch_bounds__(256)
void jacobi_kernel(float* __restrict__ A, float* __restrict__ V,
                   int n, int nrounds)
{
  __shared__ float cs[128], sn[128];
  const int t = threadIdx.x;
  for (int id = t; id < n * n; id += 256)
    V[id] = ((id / n) == (id % n)) ? 1.f : 0.f;
  __syncthreads();

  for (int rd = 0; rd < nrounds; ++rd) {
    const int r = rd % (n - 1);
    if (t < 128) {
      int p, q; rr_pair(r, t, n, p, q);
      float app = A[(size_t)p * n + p];
      float aqq = A[(size_t)q * n + q];
      float apq = A[(size_t)p * n + q];
      float c = 1.f, s = 0.f;
      if (fabsf(apq) > 1e-30f) {
        float tau = (aqq - app) / (2.f * apq);
        float tt = (tau >= 0.f ? 1.f : -1.f) / (fabsf(tau) + sqrtf(1.f + tau * tau));
        c = 1.f / sqrtf(1.f + tt * tt);
        s = tt * c;
      }
      cs[t] = c; sn[t] = s;
    }
    __syncthreads();
    // row updates: A <- J^T A (pairs write disjoint rows)
    for (int task = t; task < n * 128; task += 256) {
      int i = task >> 8, j = task & 255;
      int p, q; rr_pair(r, i, n, p, q);
      float c = cs[i], s = sn[i];
      float apj = A[(size_t)p * n + j], aqj = A[(size_t)q * n + j];
      A[(size_t)p * n + j] = c * apj - s * aqj;
      A[(size_t)q * n + j] = s * apj + c * aqj;
    }
    __syncthreads();
    // column updates: A <- A J ; V <- V J (pairs write disjoint columns)
    for (int task = t; task < n * 128; task += 256) {
      int i = task >> 8, j = task & 255;
      int p, q; rr_pair(r, i, n, p, q);
      float c = cs[i], s = sn[i];
      float ajp = A[(size_t)j * n + p], ajq = A[(size_t)j * n + q];
      A[(size_t)j * n + p] = c * ajp - s * ajq;
      A[(size_t)j * n + q] = s * ajp + c * ajq;
      float vjp = V[(size_t)j * n + p], vjq = V[(size_t)j * n + q];
      V[(size_t)j * n + p] = c * vjp - s * vjq;
      V[(size_t)j * n + q] = s * vjp + c * vjq;
    }
    __syncthreads();
  }
}

// ---------------------------------------------------------------------------
// Kernel 6: sort eigenpairs ascending (bitonic), permute eigenvector columns.
// ---------------------------------------------------------------------------
__global__ __launch_bounds__(256)
void sort_eig_kernel(const float* __restrict__ A, const float* __restrict__ V,
                     float* __restrict__ valsS, float* __restrict__ Vs, int n)
{
  __shared__ float key[256];
  __shared__ int   idx[256];
  const int t = threadIdx.x;
  key[t] = A[(size_t)t * n + t];
  idx[t] = t;
  __syncthreads();
  for (int kk = 2; kk <= n; kk <<= 1) {
    for (int j = kk >> 1; j > 0; j >>= 1) {
      int ixj = t ^ j;
      if (ixj > t) {
        bool up = ((t & kk) == 0);
        bool sw = up ? (key[t] > key[ixj]) : (key[t] < key[ixj]);
        if (sw) {
          float tk = key[t]; key[t] = key[ixj]; key[ixj] = tk;
          int ti = idx[t]; idx[t] = idx[ixj]; idx[ixj] = ti;
        }
      }
      __syncthreads();
    }
  }
  valsS[t] = key[t];
  int src = idx[t];
  for (int r = 0; r < n; ++r)
    Vs[(size_t)r * n + t] = V[(size_t)r * n + src];
}

// ---------------------------------------------------------------------------
// Block reduction helpers (blockDim.x == 256, all threads participate)
// ---------------------------------------------------------------------------
__device__ __forceinline__ float redSumF(volatile float* buf, int t, float v) {
  buf[t] = v; __syncthreads();
  for (int s = 128; s > 0; s >>= 1) { if (t < s) buf[t] += buf[t + s]; __syncthreads(); }
  float r = buf[0]; __syncthreads(); return r;
}
__device__ __forceinline__ int redSumI(volatile int* buf, int t, int v) {
  buf[t] = v; __syncthreads();
  for (int s = 128; s > 0; s >>= 1) { if (t < s) buf[t] += buf[t + s]; __syncthreads(); }
  int r = buf[0]; __syncthreads(); return r;
}
__device__ __forceinline__ int redMinI(volatile int* buf, int t, int v) {
  buf[t] = v; __syncthreads();
  for (int s = 128; s > 0; s >>= 1) {
    if (t < s && buf[t + s] < buf[t]) buf[t] = buf[t + s];
    __syncthreads();
  }
  int r = buf[0]; __syncthreads(); return r;
}

// ---------------------------------------------------------------------------
// Kernel 7: DPP sampling — faithful transcription of _dpp_sample/_gram_schmidt
// One workgroup of 256 threads; thread t doubles as row t and column t.
// ---------------------------------------------------------------------------
__global__ __launch_bounds__(256)
void dpp_kernel(const float* __restrict__ vals, const float* __restrict__ Vs,
                const float* __restrict__ u_sel, const float* __restrict__ u_item,
                float* __restrict__ Vw, float* __restrict__ subset_out, int n)
{
  __shared__ float fred[256];
  __shared__ int   ired[256];
  __shared__ float sc[256];
  __shared__ float vj[256];
  __shared__ float v0[256];
  __shared__ int   mask[256];
  __shared__ float subset[256];
  const int t = threadIdx.x;

  float val = vals[t];
  int keep = (u_sel[t] < val / (val + 1.f)) ? 1 : 0;
  mask[t] = keep;
  subset[t] = 0.f;
  __syncthreads();
  const int k = redSumI(ired, t, keep);

  {  // V0 = vecs * keep (column mask); thread t owns column t
    float m = mask[t] ? 1.f : 0.f;
    for (int r = 0; r < n; ++r)
      Vw[(size_t)r * n + t] = Vs[(size_t)r * n + t] * m;
  }
  __syncthreads();

  for (int i = 0; i < n; ++i) {
    if (i >= k) break;   // uniform: remaining iterations are no-ops in reference

    // p[t] = sum_j Vw[t][j]^2  (row sums of squares)
    float s = 0.f;
    {
      const float* row = Vw + (size_t)t * n;
      for (int j = 0; j < n; ++j) { float x = row[j]; s += x * x; }
    }
    // inclusive scan (Hillis-Steele) of p into sc, then normalize
    sc[t] = s; __syncthreads();
    for (int o = 1; o < n; o <<= 1) {
      float add = (t >= o) ? sc[t - o] : 0.f;
      __syncthreads();
      sc[t] += add;
      __syncthreads();
    }
    float total = sc[n - 1];
    float c = sc[t] / fmaxf(total, 1e-12f);

    // item = argmax(u_item[i] < c)  (first True, 0 if none)
    float ui = u_item[i];
    int item = redMinI(ired, t, (ui < c) ? t : n);
    if (item == n) item = 0;
    if (t == 0) subset[item] = 1.f;

    // j = argmax(mask & |V[item,:]| > 0)
    int jsel = redMinI(ired, t, (mask[t] && fabsf(Vw[(size_t)item * n + t]) > 0.f) ? t : n);
    if (jsel == n) jsel = 0;

    vj[t] = Vw[(size_t)t * n + jsel];
    __syncthreads();
    float d = vj[item];
    float denom = (fabsf(d) > 0.f) ? d : 1.f;

    int cnt = redSumI(ired, t, mask[t]);   // count of OLD mask
    bool many = cnt > 1;
    if (t == 0) mask[jsel] = 0;            // mask_new
    __syncthreads();

    if (many) {
      // row factors V[item,:] / denom captured before update
      sc[t] = Vw[(size_t)item * n + t] / denom;
      __syncthreads();
      {  // V = (V - outer(Vj, rowfac)) * mask_new (columns)
        float vjr = vj[t];
        float* row = Vw + (size_t)t * n;
        for (int j = 0; j < n; ++j) {
          float nv = row[j] - vjr * sc[j];
          row[j] = mask[j] ? nv : 0.f;
        }
      }
      __syncthreads();
      // Gram-Schmidt: sequential over columns a
      for (int a = 0; a < n; ++a) {
        v0[t] = Vw[(size_t)t * n + a];
        __syncthreads();
        float cf = 0.f;
        if (t < a && mask[t]) {            // coef[j] = dot(V[:,j], v0) * lt
          for (int r = 0; r < n; ++r) cf += Vw[(size_t)r * n + t] * v0[r];
        }
        sc[t] = cf;
        __syncthreads();
        float acc = 0.f;
        {
          const float* row = Vw + (size_t)t * n;
          for (int j = 0; j < a; ++j) acc += row[j] * sc[j];
        }
        float vn = v0[t] - acc;
        float nrm = sqrtf(redSumF(fred, t, vn * vn));
        Vw[(size_t)t * n + a] = mask[a] ? (vn / fmaxf(nrm, 1e-12f)) : v0[t];
        __syncthreads();
      }
    } else {
      // V_new = V * mask_new  (original V, only zero dropped columns)
      if (!mask[t]) {
        for (int r = 0; r < n; ++r) Vw[(size_t)r * n + t] = 0.f;
      }
      __syncthreads();
    }
  }

  __syncthreads();
  float sv = subset[t];
  if (k >= n) sv = 1.f;
  if (k == 0) sv = 0.f;
  subset_out[t] = sv;
}

// ---------------------------------------------------------------------------
// Prediction head (M=1 matvecs — not worth WMMA)
// ---------------------------------------------------------------------------
__global__ void pick_kernel(const float* __restrict__ subset,
                            const float* __restrict__ words,
                            float* __restrict__ pick)
{
  int c = blockIdx.x * blockDim.x + threadIdx.x;
  if (c >= 2048) return;
  float s = 0.f;
  for (int i = 0; i < 256; ++i) s += subset[i] * words[(size_t)i * 2048 + c];
  pick[c] = s;
}

__global__ void p1_kernel(const float* __restrict__ pick,
                          const float* __restrict__ w1,
                          const float* __restrict__ b1,
                          float* __restrict__ p1)
{
  int j = blockIdx.x * blockDim.x + threadIdx.x;
  if (j >= 4096) return;
  float s = b1[j];
  for (int c = 0; c < 2048; ++c) s += pick[c] * w1[(size_t)c * 4096 + j];
  p1[j] = fmaxf(s, 0.f);   // ReLU
}

__global__ __launch_bounds__(256)
void final_kernel(const float* __restrict__ p1, const float* __restrict__ w2,
                  const float* __restrict__ b2, float* __restrict__ out)
{
  __shared__ float red[256];
  int t = threadIdx.x;
  float s = 0.f;
  for (int j = t; j < 4096; j += 256) s += p1[j] * w2[j];
  red[t] = s; __syncthreads();
  for (int k = 128; k > 0; k >>= 1) { if (t < k) red[t] += red[t + k]; __syncthreads(); }
  if (t == 0) out[0] = red[0] + b2[0];
}

// ---------------------------------------------------------------------------
// Host launcher
// ---------------------------------------------------------------------------
extern "C" void kernel_launch(void* const* d_in, const int* in_sizes, int n_in,
                              void* d_out, int out_size, void* d_ws, size_t ws_size,
                              hipStream_t stream)
{
  const float* words   = (const float*)d_in[0];
  const float* context = (const float*)d_in[1];
  const float* emb_w1  = (const float*)d_in[2];
  const float* emb_b1  = (const float*)d_in[3];
  const float* emb_w2  = (const float*)d_in[4];
  const float* emb_b2  = (const float*)d_in[5];
  const float* pred_w1 = (const float*)d_in[6];
  const float* pred_b1 = (const float*)d_in[7];
  const float* pred_w2 = (const float*)d_in[8];
  const float* pred_b2 = (const float*)d_in[9];
  const float* u_sel   = (const float*)d_in[10];
  const float* u_item  = (const float*)d_in[11];
  (void)in_sizes; (void)n_in; (void)out_size; (void)ws_size;

  char* w = (char*)d_ws;
  size_t off = 0;
  auto alloc = [&](size_t bytes) -> void* {
    void* p = w + off;
    off += (bytes + 255) & ~(size_t)255;
    return p;
  };
  unsigned short* xb  = (unsigned short*)alloc((size_t)256 * 4096 * 2);  // x bf16
  unsigned short* hb  = (unsigned short*)alloc((size_t)256 * 8192 * 2);  // h bf16
  float* E      = (float*)alloc((size_t)256 * 4096 * 4);
  float* Amat   = (float*)alloc((size_t)256 * 256 * 4);  // K, then Jacobi-destroyed
  float* V      = (float*)alloc((size_t)256 * 256 * 4);
  float* Vs     = (float*)alloc((size_t)256 * 256 * 4);
  float* valsS  = (float*)alloc((size_t)256 * 4);
  float* Vw     = (float*)alloc((size_t)256 * 256 * 4);
  float* subset = (float*)alloc((size_t)256 * 4);
  float* pick   = (float*)alloc((size_t)2048 * 4);
  float* p1     = (float*)alloc((size_t)4096 * 4);

  // 1) x = concat(words, context) -> bf16
  build_x_kernel<<<4096, 256, 0, stream>>>(words, context, xb);
  // 2) h = ELU(x @ emb_w1 + b1), bf16 out
  gemm_bf16_kernel<1, 1><<<dim3(8192 / 128, 256 / 128), 128, 0, stream>>>(
      xb, emb_w1, emb_b1, hb, 256, 8192, 4096);
  // 3) E = h @ emb_w2 + b2, f32 out
  gemm_bf16_kernel<0, 0><<<dim3(4096 / 128, 256 / 128), 128, 0, stream>>>(
      hb, emb_w2, emb_b2, E, 256, 4096, 8192);
  // 4) K = E E^T (fp32 WMMA)
  syrk_f32_kernel<<<dim3(2, 2), 128, 0, stream>>>(E, Amat, 256, 4096);
  // 5) eigh: 12 Jacobi sweeps of 255 rounds
  jacobi_kernel<<<1, 256, 0, stream>>>(Amat, V, 256, 255 * 12);
  // 6) sort eigenpairs ascending
  sort_eig_kernel<<<1, 256, 0, stream>>>(Amat, V, valsS, Vs, 256);
  // 7) DPP sampling
  dpp_kernel<<<1, 256, 0, stream>>>(valsS, Vs, u_sel, u_item, Vw, subset, 256);
  // 8) prediction head
  pick_kernel<<<8, 256, 0, stream>>>(subset, words, pick);
  p1_kernel<<<16, 256, 0, stream>>>(pick, pred_w1, pred_b1, p1);
  final_kernel<<<1, 256, 0, stream>>>(p1, pred_w2, pred_b2, (float*)d_out);
}
